// BartCustomMaskAttention_43774306680882
// MI455X (gfx1250) — compile-verified, matrix operand order checked by reading, not run on hardware
//
#include <hip/hip_runtime.h>

// Problem sizes (fixed by the reference)
#define BB 4
#define TT 1024
#define EE 1024
#define HH 16
#define DD 64

typedef __attribute__((ext_vector_type(16))) __bf16 v16bf;
typedef __attribute__((ext_vector_type(8)))  __bf16 v8bf;
typedef __attribute__((ext_vector_type(8)))  float  v8f;

// ---------------------------------------------------------------------------
// WMMA helper: D = A(16x32 bf16) x B(32x16 bf16) + C(16x16 f32)
// ---------------------------------------------------------------------------
static __device__ inline v8f wmma_bf16(v16bf a, v16bf b, v8f c) {
  return __builtin_amdgcn_wmma_f32_16x16x32_bf16(
      /*neg_a=*/false, a, /*neg_b=*/false, b,
      /*c_mod=*/(short)0, c, /*reuse_a=*/false, /*reuse_b=*/false);
}

// A-operand fragment (16x32, K-chunk starting at k0) built from an f32 row.
// Per ISA layout: lane<16 holds K = k0+{0..7, 16..23}; lane>=16 holds
// K = k0+{8..15, 24..31}.  half = lane>>4.
static __device__ inline v16bf frag_from_f32(const float* __restrict__ row,
                                             int k0, int half) {
  const float* p0 = row + k0 + 8 * half;
  const float* p1 = row + k0 + 16 + 8 * half;
  float4 a0 = *(const float4*)(p0);
  float4 a1 = *(const float4*)(p0 + 4);
  float4 b0 = *(const float4*)(p1);
  float4 b1 = *(const float4*)(p1 + 4);
  v16bf f;
  f[0] = (__bf16)a0.x;  f[1] = (__bf16)a0.y;  f[2] = (__bf16)a0.z;  f[3] = (__bf16)a0.w;
  f[4] = (__bf16)a1.x;  f[5] = (__bf16)a1.y;  f[6] = (__bf16)a1.z;  f[7] = (__bf16)a1.w;
  f[8] = (__bf16)b0.x;  f[9] = (__bf16)b0.y;  f[10] = (__bf16)b0.z; f[11] = (__bf16)b0.w;
  f[12] = (__bf16)b1.x; f[13] = (__bf16)b1.y; f[14] = (__bf16)b1.z; f[15] = (__bf16)b1.w;
  return f;
}

// Same fragment but from an already-bf16 row (global or LDS): two 16B loads.
static __device__ inline v16bf frag_from_bf16(const __bf16* __restrict__ row,
                                              int k0, int half) {
  v8bf lo = *(const v8bf*)(row + k0 + 8 * half);
  v8bf hi = *(const v8bf*)(row + k0 + 16 + 8 * half);
  v16bf f;
#pragma unroll
  for (int i = 0; i < 8; ++i) { f[i] = lo[i]; f[8 + i] = hi[i]; }
  return f;
}

// ---------------------------------------------------------------------------
// Kernel 1: fused Q/K/V projection.  grid.z selects which projection.
//   X[4096,1024] f32 x W^T (+bias, Q also * D^-0.5) -> bf16
//   Q,K stored [B,H,T,D];  V stored transposed [B,H,D,T] (B-operand layout).
// Block = 256 threads (8 waves); wave tile = 32(M) x 64(N); block tile 128x128.
// ---------------------------------------------------------------------------
__global__ __launch_bounds__(256) void qkv_proj_kernel(
    const float* __restrict__ X,
    const float* __restrict__ Wq, const float* __restrict__ bq,
    const float* __restrict__ Wk, const float* __restrict__ bk,
    const float* __restrict__ Wv, const float* __restrict__ bv,
    __bf16* __restrict__ Qb, __bf16* __restrict__ Kb, __bf16* __restrict__ Vt) {
  const int z = blockIdx.z;
  const float* W    = (z == 0) ? Wq : (z == 1) ? Wk : Wv;
  const float* bias = (z == 0) ? bq : (z == 1) ? bk : bv;
  const float scale = (z == 0) ? 0.125f : 1.0f;  // D^-0.5 = 1/8

  const int wave = threadIdx.x >> 5;
  const int lane = threadIdx.x & 31;
  const int half = lane >> 4;
  const int lr   = lane & 15;

  const int m0 = blockIdx.x * 128 + (wave >> 1) * 32;
  const int n0 = blockIdx.y * 128 + (wave & 1) * 64;

  v8f acc[2][4] = {};

  for (int k0 = 0; k0 < EE; k0 += 32) {
    v16bf a[2], b[4];
    a[0] = frag_from_f32(X + (size_t)(m0 + lr) * EE, k0, half);
    a[1] = frag_from_f32(X + (size_t)(m0 + 16 + lr) * EE, k0, half);
#pragma unroll
    for (int jj = 0; jj < 4; ++jj)
      b[jj] = frag_from_f32(W + (size_t)(n0 + 16 * jj + lr) * EE, k0, half);
#pragma unroll
    for (int ii = 0; ii < 2; ++ii)
#pragma unroll
      for (int jj = 0; jj < 4; ++jj)
        acc[ii][jj] = wmma_bf16(a[ii], b[jj], acc[ii][jj]);
  }

  // Epilogue.  C layout: VGPR p, lane l -> row M = p + 8*(l>>4), col N = l&15.
#pragma unroll
  for (int ii = 0; ii < 2; ++ii) {
#pragma unroll
    for (int jj = 0; jj < 4; ++jj) {
      const int n  = n0 + 16 * jj + lr;
      const int h  = n >> 6;
      const int d  = n & 63;
      const float bv_ = bias[n];
      const int mbase = m0 + 16 * ii + 8 * half;   // 8 consecutive rows per lane
      const int bidx  = mbase >> 10;               // batch (rows never straddle)
      const int tbase = mbase & 1023;
      if (z < 2) {
        __bf16* O = (z == 0) ? Qb : Kb;
#pragma unroll
        for (int p = 0; p < 8; ++p) {
          float v = (acc[ii][jj][p] + bv_) * scale;
          O[(((size_t)bidx * HH + h) * TT + (tbase + p)) * DD + d] = (__bf16)v;
        }
      } else {
        // V transposed: lane's 8 consecutive t values -> one packed 16B store.
        v8bf pk;
#pragma unroll
        for (int p = 0; p < 8; ++p) pk[p] = (__bf16)(acc[ii][jj][p] + bv_);
        *(v8bf*)(Vt + (((size_t)bidx * HH + h) * DD + d) * TT + tbase) = pk;
      }
    }
  }
}

// ---------------------------------------------------------------------------
// Kernel 2: attention for one (b, h, 16-row tile of t).
//   Phase 1: S = Q K^T + mask  -> LDS f32 [16][1024]   (WMMA)
//   Phase 2: softmax + relation mask -> LDS bf16 P [16][1024]
//   Phase 3: P V  (V pre-transposed) -> AO [B,T,E] bf16  (WMMA, k-split)
// LDS: 64KB + 32KB = 96KB.
// ---------------------------------------------------------------------------
__global__ __launch_bounds__(256) void attention_kernel(
    const __bf16* __restrict__ Qb, const __bf16* __restrict__ Kb,
    const __bf16* __restrict__ Vt,
    const float* __restrict__ amask, const int* __restrict__ rel,
    const float* __restrict__ heads_mask,
    __bf16* __restrict__ AO) {
  __shared__ float  S[16 * 1024];   // scores (f32), later reused as reduce scratch
  __shared__ __bf16 P[16 * 1024];   // probabilities (bf16)

  const int t0 = blockIdx.x * 16;
  const int h  = blockIdx.y;
  const int bz = blockIdx.z;
  const int bh = bz * HH + h;

  const int wave = threadIdx.x >> 5;
  const int lane = threadIdx.x & 31;
  const int half = lane >> 4;
  const int lr   = lane & 15;

  // --- Phase 1: scores -----------------------------------------------------
  const __bf16* Qrow = Qb + ((size_t)bh * TT + t0 + lr) * DD;
  const v16bf a0 = frag_from_bf16(Qrow, 0, half);
  const v16bf a1 = frag_from_bf16(Qrow, 32, half);
  const float* mbase = amask + (size_t)bz * TT * TT;

  for (int j = wave * 8; j < wave * 8 + 8; ++j) {
    const int s0 = j * 16;
    const __bf16* Krow = Kb + ((size_t)bh * TT + s0 + lr) * DD;
    v16bf b0 = frag_from_bf16(Krow, 0, half);
    v16bf b1 = frag_from_bf16(Krow, 32, half);
    v8f acc = {};
    acc = wmma_bf16(a0, b0, acc);
    acc = wmma_bf16(a1, b1, acc);
    const int s = s0 + lr;
#pragma unroll
    for (int p = 0; p < 8; ++p) {
      const int tl = p + 8 * half;
      S[tl * 1024 + s] = acc[p] + mbase[(size_t)(t0 + tl) * TT + s];
    }
  }
  __syncthreads();

  // --- Phase 2: softmax + relation mask -----------------------------------
  {
    const int r = threadIdx.x >> 4;   // row 0..15
    const int q = threadIdx.x & 15;   // 16 threads per row, 64 cols each
    float* Srow = S + r * 1024;
    float mx = -3.4e38f;
    for (int c = q * 64; c < q * 64 + 64; ++c) mx = fmaxf(mx, Srow[c]);
#pragma unroll
    for (int m = 1; m < 16; m <<= 1) mx = fmaxf(mx, __shfl_xor(mx, m, 32));
    float sum = 0.f;
    for (int c = q * 64; c < q * 64 + 64; ++c) {
      float e = __expf(Srow[c] - mx);
      Srow[c] = e;
      sum += e;
    }
#pragma unroll
    for (int m = 1; m < 16; m <<= 1) sum += __shfl_xor(sum, m, 32);
    const float inv = 1.0f / sum;
    const float hm = heads_mask[h];
    __bf16* Prow = P + r * 1024;
    if (hm != 0.f) {   // relation-indexed head(s) only: avoids rel traffic elsewhere
      const int* relrow = rel + ((size_t)bz * TT + (t0 + r)) * TT;
      const float base = 1.0f - hm;
      for (int c = q * 64; c < q * 64 + 64; ++c) {
        float pv = Srow[c] * inv;
        pv *= base + hm * (relrow[c] > 0 ? 1.0f : 0.0f);
        Prow[c] = (__bf16)pv;
      }
    } else {
      for (int c = q * 64; c < q * 64 + 64; ++c)
        Prow[c] = (__bf16)(Srow[c] * inv);
    }
  }
  __syncthreads();

  // --- Phase 3: out = P @ V  (k-split: waves w and w+4 share tile w&3) -----
  {
    const int j  = wave & 3;   // d tile (16 cols)
    const int kh = wave >> 2;  // k half
    const __bf16* Prow = P + lr * 1024;
    const __bf16* Vrow = Vt + ((size_t)bh * DD + j * 16 + lr) * TT;
    v8f acc = {};
    for (int k0 = kh * 512; k0 < kh * 512 + 512; k0 += 32) {
      v16bf a = frag_from_bf16(Prow, k0, half);
      v16bf b = frag_from_bf16(Vrow, k0, half);
      acc = wmma_bf16(a, b, acc);
    }
    __syncthreads();                     // S no longer needed; reuse as scratch
    *(v8f*)(S + wave * 256 + lane * 8) = acc;
    __syncthreads();
    if (wave < 4) {
      v8f x = *(v8f*)(S + wave * 256 + lane * 8);
      v8f y = *(v8f*)(S + (wave + 4) * 256 + lane * 8);
      const int d = wave * 16 + lr;
#pragma unroll
      for (int p = 0; p < 8; ++p) {
        const int t = t0 + p + 8 * half;
        AO[((size_t)bz * TT + t) * EE + h * DD + d] = (__bf16)(x[p] + y[p]);
      }
    }
  }
}

// ---------------------------------------------------------------------------
// Kernel 3: output projection.  AO[4096,1024] bf16 x o_w^T + o_b -> f32 out.
// Same tiling as kernel 1; f32 stores are 64B-coalesced per C-VGPR.
// ---------------------------------------------------------------------------
__global__ __launch_bounds__(256) void out_proj_kernel(
    const __bf16* __restrict__ A,
    const float* __restrict__ Wo, const float* __restrict__ bo,
    float* __restrict__ Out) {
  const int wave = threadIdx.x >> 5;
  const int lane = threadIdx.x & 31;
  const int half = lane >> 4;
  const int lr   = lane & 15;

  const int m0 = blockIdx.x * 128 + (wave >> 1) * 32;
  const int n0 = blockIdx.y * 128 + (wave & 1) * 64;

  v8f acc[2][4] = {};
  for (int k0 = 0; k0 < EE; k0 += 32) {
    v16bf a[2], b[4];
    a[0] = frag_from_bf16(A + (size_t)(m0 + lr) * EE, k0, half);
    a[1] = frag_from_bf16(A + (size_t)(m0 + 16 + lr) * EE, k0, half);
#pragma unroll
    for (int jj = 0; jj < 4; ++jj)
      b[jj] = frag_from_f32(Wo + (size_t)(n0 + 16 * jj + lr) * EE, k0, half);
#pragma unroll
    for (int ii = 0; ii < 2; ++ii)
#pragma unroll
      for (int jj = 0; jj < 4; ++jj)
        acc[ii][jj] = wmma_bf16(a[ii], b[jj], acc[ii][jj]);
  }
#pragma unroll
  for (int ii = 0; ii < 2; ++ii) {
#pragma unroll
    for (int jj = 0; jj < 4; ++jj) {
      const int n = n0 + 16 * jj + lr;
      const float bv = bo[n];
#pragma unroll
      for (int p = 0; p < 8; ++p) {
        const int m = m0 + 16 * ii + p + 8 * half;
        Out[(size_t)m * EE + n] = acc[ii][jj][p] + bv;
      }
    }
  }
}

// ---------------------------------------------------------------------------
// Launch: workspace = Qb(8MB) | Kb(8MB) | Vt(8MB) | AO(8MB) = 32MB bf16.
// ---------------------------------------------------------------------------
extern "C" void kernel_launch(void* const* d_in, const int* in_sizes, int n_in,
                              void* d_out, int out_size, void* d_ws, size_t ws_size,
                              hipStream_t stream) {
  const float* hidden = (const float*)d_in[0];
  const float* amask  = (const float*)d_in[1];
  const int*   rel    = (const int*)d_in[2];
  const float* hmask  = (const float*)d_in[3];
  const float* q_w = (const float*)d_in[4];
  const float* q_b = (const float*)d_in[5];
  const float* k_w = (const float*)d_in[6];
  const float* k_b = (const float*)d_in[7];
  const float* v_w = (const float*)d_in[8];
  const float* v_b = (const float*)d_in[9];
  const float* o_w = (const float*)d_in[10];
  const float* o_b = (const float*)d_in[11];
  float* out = (float*)d_out;

  const size_t elems = (size_t)BB * HH * TT * DD;  // 4M elements per buffer
  __bf16* Qb = (__bf16*)d_ws;
  __bf16* Kb = Qb + elems;
  __bf16* Vt = Kb + elems;
  __bf16* AO = Vt + elems;

  // 1) QKV projections (z = 0:Q, 1:K, 2:V-transposed)
  qkv_proj_kernel<<<dim3(32, 8, 3), 256, 0, stream>>>(
      hidden, q_w, q_b, k_w, k_b, v_w, v_b, Qb, Kb, Vt);

  // 2) attention per (t-tile, head, batch)
  attention_kernel<<<dim3(TT / 16, HH, BB), 256, 0, stream>>>(
      Qb, Kb, Vt, amask, rel, hmask, AO);

  // 3) output projection
  out_proj_kernel<<<dim3(32, 8, 1), 256, 0, stream>>>(AO, o_w, o_b, out);
}
